// MoELayer_22187801051301
// MI455X (gfx1250) — compile-verified, matrix operand order checked by reading, not run on hardware
//
#include <hip/hip_runtime.h>
#include <hip/hip_bf16.h>
#include <math.h>

// ---------------------------------------------------------------------------
// MoE layer (top-1 routing), MI455X / gfx1250, wave32 + WMMA bf16 path.
// Packed bf16 weight panels + 4-wide n-tile register blocking:
// hot loop = 4x ds_load_b128 (A) + 8x global_load_b128 (B) per 8 WMMAs.
// ---------------------------------------------------------------------------

typedef __attribute__((ext_vector_type(16))) __bf16 v16bf;
typedef __attribute__((ext_vector_type(8)))  __bf16 v8bf;
typedef __attribute__((ext_vector_type(4)))  __bf16 v4bf;
typedef __attribute__((ext_vector_type(8)))  float  v8f;

// Problem sizes (fixed by the reference)
constexpr int Bb = 8;
constexpr int Nn = 2048;
constexpr int Hh = 1024;
constexpr int Ee = 8;
constexpr int Dd = 2048;
constexpr int T  = Bb * Nn;                           // 16384 tokens
constexpr int TILE_M    = 32;                         // tokens per FFN tile
constexpr int NTILES    = (T + TILE_M - 1) / TILE_M;  // 512
constexpr int RB_BLOCKS = T / 64;                     // router blocks = 256

// Packed-weight geometry: panels of (k-block=32) x (n-tile=16) = 512 bf16 (1KB)
constexpr int KB1 = Hh / 32;   // 32  k-blocks, GEMM1
constexpr int NT1 = Dd / 16;   // 128 n-tiles,  GEMM1
constexpr int KB2 = Dd / 32;   // 64  k-blocks, GEMM2
constexpr int NT2 = Hh / 16;   // 64  n-tiles,  GEMM2

// LDS strides (bf16 elems); multiple of 8 => 16B-aligned ds_load_b128 frags.
constexpr int XS_STRIDE = 1040;
constexpr int HS_STRIDE = 2064;

// Workspace layout (bytes)
constexpr size_t WS_CNT    = 0;                                   // 8 ints
constexpr size_t WS_BUCKET = 128;                                 // E*T ints
constexpr size_t WS_PSUM   = WS_BUCKET + (size_t)Ee * T * 4;      // 256*8 f32
constexpr size_t WS_ENT    = WS_PSUM + (size_t)RB_BLOCKS * Ee * 4;// 256 f32
constexpr size_t WS_W1B    = (WS_ENT + (size_t)RB_BLOCKS * 4 + 4095) & ~(size_t)4095;
constexpr size_t W1B_BYTES = (size_t)Ee * Hh * Dd * 2;            // 33.5 MB
constexpr size_t WS_W2B    = WS_W1B + W1B_BYTES;
constexpr size_t W2B_BYTES = (size_t)Ee * Dd * Hh * 2;            // 33.5 MB

// ---------------------------------------------------------------------------
// WMMA fragment layout (16-bit operands, ISA 7.12.2):
//   lanes 0-15 : row/col = lane,    K = {0..7, 16..23} (+k_base)
//   lanes 16-31: row/col = lane-16, K = {8..15, 24..31} (+k_base)
// ---------------------------------------------------------------------------

__device__ inline v16bf load_a_frag_lds(const __bf16* __restrict__ S, int stride,
                                        int m_base, int kb, int lane) {
    int m = m_base + (lane & 15);
    const __bf16* p = S + (size_t)m * stride + kb + ((lane < 16) ? 0 : 8);
    v8bf lo = *(const v8bf*)(p);        // ds_load_b128
    v8bf hi = *(const v8bf*)(p + 16);   // ds_load_b128
    return __builtin_shufflevector(lo, hi, 0, 1, 2, 3, 4, 5, 6, 7,
                                           8, 9, 10, 11, 12, 13, 14, 15);
}

// Packed-B panel (512 bf16 per (k-block, n-tile)); `p` pre-offset by lane*8.
__device__ inline v16bf load_b_frag_pk(const __bf16* __restrict__ p) {
    v8bf lo = *(const v8bf*)(p);        // global_load_b128 (coalesced)
    v8bf hi = *(const v8bf*)(p + 256);  // global_load_b128 (+512B)
    return __builtin_shufflevector(lo, hi, 0, 1, 2, 3, 4, 5, 6, 7,
                                           8, 9, 10, 11, 12, 13, 14, 15);
}

// ---------------------------------------------------------------------------
// Kernel W: repack f32 [E,K,N] weights -> bf16 WMMA-B fragment panels.
// One thread per (e, n-tile, k-block, lane): reads 16 f32, writes 32B.
// ---------------------------------------------------------------------------
__global__ __launch_bounds__(256) void moe_pack_weights_kernel(
    const float* __restrict__ W, __bf16* __restrict__ Wb, int K, int N) {
    const int tid  = blockIdx.x * blockDim.x + threadIdx.x;
    const int lane = tid & 31;
    const int g    = tid >> 5;
    const int KBLK = K >> 5;
    const int NT   = N >> 4;
    const int kb   = g % KBLK;
    const int cg   = g / KBLK;          // e*NT + nt
    const int nt   = cg % NT;
    const int e    = cg / NT;
    if (e >= Ee) return;

    const float* __restrict__ src = W + (size_t)e * K * N;
    const int n  = nt * 16 + (lane & 15);
    const int k0 = kb * 32 + ((lane < 16) ? 0 : 8);

    v8bf lo, hi;
#pragma unroll
    for (int i = 0; i < 8; ++i) {
        lo[i] = (__bf16)src[(size_t)(k0 + i) * N + n];
        hi[i] = (__bf16)src[(size_t)(k0 + 16 + i) * N + n];
    }
    const size_t dbase = ((size_t)cg * KBLK + kb) * 512;
    *(v8bf*)(Wb + dbase + (size_t)lane * 8)       = lo;
    *(v8bf*)(Wb + dbase + 256 + (size_t)lane * 8) = hi;
}

// ---------------------------------------------------------------------------
// Kernel 0: zero bucket counters.
// ---------------------------------------------------------------------------
__global__ void moe_init_kernel(int* __restrict__ cnt) {
    if (threadIdx.x < Ee) cnt[threadIdx.x] = 0;
}

// ---------------------------------------------------------------------------
// Kernel 1: router. One wave per token; Wr cached in LDS. Deterministic
// per-block loss partials; token->expert bucket append via int atomics.
// ---------------------------------------------------------------------------
__global__ __launch_bounds__(256) void moe_router_kernel(
    const float* __restrict__ x, const float* __restrict__ Wr,
    const float* __restrict__ br, int* __restrict__ cnt,
    int* __restrict__ bucket, float* __restrict__ blkPsum,
    float* __restrict__ blkEnt) {
    __shared__ float sWr[Hh * Ee];      // 32 KB
    __shared__ float sWaveP[8][Ee];
    __shared__ float sWaveEnt[8];

    for (int i = threadIdx.x; i < Hh * Ee; i += blockDim.x) sWr[i] = Wr[i];
    __syncthreads();

    const int wave = threadIdx.x >> 5;
    const int lane = threadIdx.x & 31;
    const int base = blockIdx.x * 64;

    float pAcc[Ee];
#pragma unroll
    for (int e = 0; e < Ee; ++e) pAcc[e] = 0.0f;
    float entAcc = 0.0f;

    for (int it = 0; it < 8; ++it) {
        const int t = base + it * 8 + wave;
        const float* __restrict__ xr = x + (size_t)t * Hh;

        float acc[Ee];
#pragma unroll
        for (int e = 0; e < Ee; ++e) acc[e] = 0.0f;
        for (int h = lane; h < Hh; h += 32) {
            const float xv = xr[h];
            const float* wrow = &sWr[h * Ee];
#pragma unroll
            for (int e = 0; e < Ee; ++e) acc[e] = fmaf(xv, wrow[e], acc[e]);
        }
#pragma unroll
        for (int e = 0; e < Ee; ++e) {
#pragma unroll
            for (int m = 16; m >= 1; m >>= 1)
                acc[e] += __shfl_xor(acc[e], m, 32);
        }

        if (lane == 0) {
            float logit[Ee];
            float mx = -3.4e38f;
#pragma unroll
            for (int e = 0; e < Ee; ++e) {
                logit[e] = acc[e] + br[e];
                mx = fmaxf(mx, logit[e]);
            }
            float p[Ee];
            float se = 0.0f;
#pragma unroll
            for (int e = 0; e < Ee; ++e) { p[e] = __expf(logit[e] - mx); se += p[e]; }
            const float inv = 1.0f / se;
            float ent = 0.0f;
            int   best = 0;
            float bp = -1.0f;
#pragma unroll
            for (int e = 0; e < Ee; ++e) {
                p[e] *= inv;
                ent -= p[e] * logf(p[e] + 1e-8f);
                pAcc[e] += p[e];
                if (p[e] > bp) { bp = p[e]; best = e; }   // first-max tiebreak
            }
            entAcc += ent;
            const int pos = atomicAdd(&cnt[best], 1);
            bucket[best * T + pos] = t;
        }
    }

    if (lane == 0) {
#pragma unroll
        for (int e = 0; e < Ee; ++e) sWaveP[wave][e] = pAcc[e];
        sWaveEnt[wave] = entAcc;
    }
    __syncthreads();
    if (threadIdx.x == 0) {
        float ps[Ee];
#pragma unroll
        for (int e = 0; e < Ee; ++e) ps[e] = 0.0f;
        float es = 0.0f;
        for (int w = 0; w < 8; ++w) {   // fixed order -> deterministic
#pragma unroll
            for (int e = 0; e < Ee; ++e) ps[e] += sWaveP[w][e];
            es += sWaveEnt[w];
        }
#pragma unroll
        for (int e = 0; e < Ee; ++e) blkPsum[blockIdx.x * Ee + e] = ps[e];
        blkEnt[blockIdx.x] = es;
    }
}

// ---------------------------------------------------------------------------
// Kernel 2: grouped expert FFN on 32-token tiles, packed bf16 weights.
//   GEMM1 (M32,K1024,N2048) -> bias -> erf GELU -> Hs (bf16, LDS)
//   GEMM2 (M32,K2048,N1024) -> bias -> +residual -> scatter store
// Register blocking: 4 n-tiles x 2 m-tiles = 8 accumulators (64 VGPRs);
// A frags loaded once per k-block, B panels for 4 n-tiles are contiguous.
// ---------------------------------------------------------------------------
__global__ __launch_bounds__(256) void moe_ffn_kernel(
    const float* __restrict__ x, const __bf16* __restrict__ W1b,
    const float* __restrict__ b1, const __bf16* __restrict__ W2b,
    const float* __restrict__ b2, const int* __restrict__ cnt,
    const int* __restrict__ bucket, float* __restrict__ out) {
    extern __shared__ char smem[];
    int*    sTok = (int*)smem;                                   // 32 ints
    __bf16* Xs   = (__bf16*)(smem + 128);                        // 32 x XS_STRIDE
    __bf16* Hs   = (__bf16*)(smem + 128 + TILE_M * XS_STRIDE * 2);

    const int e     = blockIdx.x >> 9;      // / 512
    const int tile  = blockIdx.x & 511;
    const int cn    = cnt[e];
    const int tbase = tile * TILE_M;
    if (tbase >= cn) return;

    if (threadIdx.x < TILE_M) {
        const int idx = tbase + threadIdx.x;
        sTok[threadIdx.x] = (idx < cn) ? bucket[e * T + idx] : -1;
    }
    __syncthreads();

    // ---- stage X tile f32 -> bf16 into LDS ----
    {
        const int row = threadIdx.x >> 3;   // 0..31
        const int seg = threadIdx.x & 7;    // 128 cols each
        const int t   = sTok[row];
        __bf16* dst = Xs + (size_t)row * XS_STRIDE + seg * 128;
        if (t >= 0) {
            const float4* src = (const float4*)(x + (size_t)t * Hh + seg * 128);
#pragma unroll
            for (int i = 0; i < 32; ++i) {
                const float4 v = src[i];
                v4bf w = {(__bf16)v.x, (__bf16)v.y, (__bf16)v.z, (__bf16)v.w};
                *(v4bf*)(dst + i * 4) = w;
            }
        } else {
            const v4bf z = {(__bf16)0.0f, (__bf16)0.0f, (__bf16)0.0f, (__bf16)0.0f};
#pragma unroll
            for (int i = 0; i < 32; ++i) *(v4bf*)(dst + i * 4) = z;
        }
    }
    __syncthreads();

    const int wave = threadIdx.x >> 5;
    const int lane = threadIdx.x & 31;

    // ---- GEMM1 + GELU -> Hs : 4 groups of 4 n-tiles ----
    const float* __restrict__ b1e = b1 + (size_t)e * Dd;
    for (int g = 0; g < 4; ++g) {
        const int     nt0 = wave * 16 + g * 4;                  // first n-tile
        const __bf16* pb  = W1b + ((size_t)(e * NT1 + nt0) * KB1) * 512
                                + (size_t)lane * 8;
        v8f acc[4][2] = {};
#pragma unroll 2
        for (int kbi = 0; kbi < KB1; ++kbi) {
            const v16bf a0 = load_a_frag_lds(Xs, XS_STRIDE, 0,  kbi * 32, lane);
            const v16bf a1 = load_a_frag_lds(Xs, XS_STRIDE, 16, kbi * 32, lane);
#pragma unroll
            for (int j = 0; j < 4; ++j) {
                const v16bf b = load_b_frag_pk(pb + (size_t)j * (KB1 * 512)
                                                  + (size_t)kbi * 512);
                acc[j][0] = __builtin_amdgcn_wmma_f32_16x16x32_bf16(
                    false, a0, false, b, (short)0, acc[j][0], false, false);
                acc[j][1] = __builtin_amdgcn_wmma_f32_16x16x32_bf16(
                    false, a1, false, b, (short)0, acc[j][1], false, false);
            }
        }
        const int moff = (lane < 16) ? 0 : 8;
#pragma unroll
        for (int j = 0; j < 4; ++j) {
            const int   n    = (nt0 + j) * 16 + (lane & 15);
            const float bias = b1e[n];
#pragma unroll
            for (int r = 0; r < 8; ++r) {
                const float v0 = acc[j][0][r] + bias;
                const float v1 = acc[j][1][r] + bias;
                const float g0 = 0.5f * v0 * (1.0f + erff(v0 * 0.70710678118654752f));
                const float g1 = 0.5f * v1 * (1.0f + erff(v1 * 0.70710678118654752f));
                Hs[(size_t)(moff + r) * HS_STRIDE + n]      = (__bf16)g0;
                Hs[(size_t)(moff + r + 16) * HS_STRIDE + n] = (__bf16)g1;
            }
        }
    }
    __syncthreads();

    // ---- GEMM2 + bias + residual -> out : 2 groups of 4 n-tiles ----
    const float* __restrict__ b2e = b2 + (size_t)e * Hh;
    for (int g = 0; g < 2; ++g) {
        const int     nt0 = wave * 8 + g * 4;
        const __bf16* pb  = W2b + ((size_t)(e * NT2 + nt0) * KB2) * 512
                                + (size_t)lane * 8;
        v8f acc[4][2] = {};
#pragma unroll 2
        for (int kbi = 0; kbi < KB2; ++kbi) {
            const v16bf a0 = load_a_frag_lds(Hs, HS_STRIDE, 0,  kbi * 32, lane);
            const v16bf a1 = load_a_frag_lds(Hs, HS_STRIDE, 16, kbi * 32, lane);
#pragma unroll
            for (int j = 0; j < 4; ++j) {
                const v16bf b = load_b_frag_pk(pb + (size_t)j * (KB2 * 512)
                                                  + (size_t)kbi * 512);
                acc[j][0] = __builtin_amdgcn_wmma_f32_16x16x32_bf16(
                    false, a0, false, b, (short)0, acc[j][0], false, false);
                acc[j][1] = __builtin_amdgcn_wmma_f32_16x16x32_bf16(
                    false, a1, false, b, (short)0, acc[j][1], false, false);
            }
        }
        const int moff = (lane < 16) ? 0 : 8;
#pragma unroll
        for (int j = 0; j < 4; ++j) {
            const int   n    = (nt0 + j) * 16 + (lane & 15);
            const float bias = b2e[n];
#pragma unroll
            for (int r = 0; r < 8; ++r) {
                const int t0 = sTok[moff + r];
                const int t1 = sTok[moff + r + 16];
                if (t0 >= 0) {
                    const size_t o = (size_t)t0 * Hh + n;
                    out[o] = x[o] + acc[j][0][r] + bias;
                }
                if (t1 >= 0) {
                    const size_t o = (size_t)t1 * Hh + n;
                    out[o] = x[o] + acc[j][1][r] + bias;
                }
            }
        }
    }
}

// ---------------------------------------------------------------------------
// Kernel 3: fixed-order reduction of router partials -> losses + counts.
// d_out tail layout: [balance_loss, entropy_loss, tokens_per_expert[8]]
// ---------------------------------------------------------------------------
__global__ void moe_finalize_kernel(const int* __restrict__ cnt,
                                    const float* __restrict__ blkPsum,
                                    const float* __restrict__ blkEnt,
                                    float* __restrict__ out_tail) {
    if (threadIdx.x == 0) {
        float ps[Ee];
#pragma unroll
        for (int e = 0; e < Ee; ++e) ps[e] = 0.0f;
        float es = 0.0f;
        for (int bidx = 0; bidx < RB_BLOCKS; ++bidx) {
#pragma unroll
            for (int e = 0; e < Ee; ++e) ps[e] += blkPsum[bidx * Ee + e];
            es += blkEnt[bidx];
        }
        float bal = 0.0f;
#pragma unroll
        for (int e = 0; e < Ee; ++e) {
            const float pi = ps[e] / (float)T;
            bal += pi * pi;
        }
        out_tail[0] = (float)Ee * bal;
        out_tail[1] = es / (float)T;
    }
    if (threadIdx.x < Ee) out_tail[2 + threadIdx.x] = (float)cnt[threadIdx.x];
}

// ---------------------------------------------------------------------------
extern "C" void kernel_launch(void* const* d_in, const int* in_sizes, int n_in,
                              void* d_out, int out_size, void* d_ws, size_t ws_size,
                              hipStream_t stream) {
    const float* x  = (const float*)d_in[0];
    const float* Wr = (const float*)d_in[1];
    const float* br = (const float*)d_in[2];
    const float* W1 = (const float*)d_in[3];
    const float* b1 = (const float*)d_in[4];
    const float* W2 = (const float*)d_in[5];
    const float* b2 = (const float*)d_in[6];
    float* out = (float*)d_out;

    char* ws = (char*)d_ws;
    int*    cnt     = (int*)(ws + WS_CNT);
    int*    bucket  = (int*)(ws + WS_BUCKET);
    float*  blkPsum = (float*)(ws + WS_PSUM);
    float*  blkEnt  = (float*)(ws + WS_ENT);
    __bf16* W1b     = (__bf16*)(ws + WS_W1B);
    __bf16* W2b     = (__bf16*)(ws + WS_W2B);

    moe_init_kernel<<<1, 32, 0, stream>>>(cnt);

    // Pack weights to bf16 fragment panels (~200 MB of L2/HBM traffic, ~9us).
    {
        const int thr1 = Ee * NT1 * KB1 * 32;   // 1,048,576
        const int thr2 = Ee * NT2 * KB2 * 32;   // 1,048,576
        moe_pack_weights_kernel<<<thr1 / 256, 256, 0, stream>>>(W1, W1b, Hh, Dd);
        moe_pack_weights_kernel<<<thr2 / 256, 256, 0, stream>>>(W2, W2b, Dd, Hh);
    }

    moe_router_kernel<<<RB_BLOCKS, 256, 0, stream>>>(x, Wr, br, cnt, bucket,
                                                     blkPsum, blkEnt);

    const size_t smem = 128 + (size_t)TILE_M * XS_STRIDE * 2
                            + (size_t)TILE_M * HS_STRIDE * 2;  // ~198.8 KB
    moe_ffn_kernel<<<Ee * NTILES, 256, smem, stream>>>(x, W1b, b1, W2b, b2,
                                                       cnt, bucket, out);

    moe_finalize_kernel<<<1, 32, 0, stream>>>(cnt, blkPsum, blkEnt,
                                              out + (size_t)Bb * Nn * Hh);
}
